// FrequencySpatialFusion_69561290326542
// MI455X (gfx1250) — matrix operand order repaired and checked by reading, not compiled, not run on hardware
//
#include <hip/hip_runtime.h>
#include <hip/hip_bf16.h>

typedef __bf16 bf16;
typedef __attribute__((ext_vector_type(16))) __bf16 v16bf;
typedef __attribute__((ext_vector_type(8)))  __bf16 bf16x8;
typedef __attribute__((ext_vector_type(8)))  float  v8f;

#define DEV static __device__ __forceinline__
#define PI_F 3.14159265358979323846f

DEV bf16  f2b(float f){ return (bf16)f; }
DEV float b2f(bf16 h){ return (float)h; }
DEV float gelu_exact(float x){ return 0.5f*x*(1.0f+erff(x*0.70710678118654752f)); }

DEV v16bf wmma_bf16(v16bf a, v16bf b, v8f& acc){ // helper just for brevity at call sites
  acc = __builtin_amdgcn_wmma_f32_16x16x32_bf16(false,a,false,b,(short)0,acc,false,false);
  return a;
}

// ---------------------------------------------------------------- basis
// dhb[32][64] bf16 (k-major), dhtb[64][32] bf16 (h-major), ph[8][32] f32 pooling matrix
__global__ void __launch_bounds__(256) k_basis(bf16* __restrict__ dhb, bf16* __restrict__ dhtb,
                                               float* __restrict__ ph){
  __shared__ float s[32*64];
  int t=threadIdx.x;
  const float scale=0.17677669529663687f; // sqrt(2/64)
  for(int i=t;i<32*64;i+=256){
    int k=i>>6,h=i&63;
    float v=cosf(PI_F*(2.0f*h+1.0f)*(float)k/128.0f)*scale;
    if(k==0) v*=0.70710678118654752f;
    s[i]=v; dhb[i]=f2b(v); dhtb[h*32+k]=f2b(v);
  }
  __syncthreads();
  for(int i=t;i<8*32;i+=256){
    int p=i>>5,k=i&31;
    float a=0.f;
    for(int j=0;j<8;j++) a+=s[k*64+p*8+j];
    ph[i]=a*0.125f;
  }
}

// ---------------------------------------------------------------- weight conversion
__global__ void __launch_bounds__(256) k_f2b(const float* __restrict__ src, bf16* __restrict__ dst, long n){
  long i=(long)blockIdx.x*256+threadIdx.x;
  if(i<n) dst[i]=f2b(src[i]);
}
// 3x3 weights: OIHW [co][ci][3][3] -> A[co][tap*256+ci]  (tap-major K so NHWC B stays contiguous)
__global__ void __launch_bounds__(256) k_wperm(const float* __restrict__ src, bf16* __restrict__ dst, long n){
  long i=(long)blockIdx.x*256+threadIdx.x;
  if(i>=n) return;
  long co=i/2304; int r=(int)(i-co*2304); int tap=r>>8, ci=r&255;
  dst[i]=f2b(src[co*2304 + (long)ci*9 + tap]);
}

// ---------------------------------------------------------------- diff: NCHW f32 + NHWC bf16
__global__ void __launch_bounds__(256) k_diff(const float* __restrict__ x1, const float* __restrict__ x2,
                                              float* __restrict__ df, bf16* __restrict__ db, long n){
  long i=(long)blockIdx.x*256+threadIdx.x;
  if(i>=n) return;
  float v=x1[i]-x2[i]; df[i]=v;
  long hw=i&4095; int c=(int)((i>>12)&255); long b=i>>20;
  db[((b<<12)+hw)*256 + c]=f2b(v);
}

// ---------------------------------------------------------------- DCT low-block, WMMA version
// per (b,c): F = Dh32 X Dh32^T ; high_in(NHWC bf16) = X - Dh32^T F Dh32 ; lowp(NHWC bf16) = Ph F Ph^T
__global__ void __launch_bounds__(256) k_dct(const float* __restrict__ diff /*NCHW*/,
                                             const bf16* __restrict__ dhb, const bf16* __restrict__ dhtb,
                                             const float* __restrict__ ph,
                                             bf16* __restrict__ high_in /*NHWC*/,
                                             bf16* __restrict__ lowp /*NHWC [b][64][256]*/){
  __shared__ float Xf[64*64];
  __shared__ bf16  Xt[64*64];   // [w][h]
  __shared__ bf16  T [32*64];   // [k][w]
  __shared__ bf16  Ft[32*32];   // [l][k]
  __shared__ bf16  U [64*32];   // [h][l]
  __shared__ float V [8*32];

  const int bc=blockIdx.x, tid=threadIdx.x;
  const int lane=tid&31, wave=tid>>5, l15=lane&15, kh=lane>>4;
  const int bb_=bc>>8, cc=bc&255;
  const float* xp=diff+(size_t)bc*4096;

  for(int i=tid;i<4096;i+=256){ float v=xp[i]; Xf[i]=v; Xt[(i&63)*64+(i>>6)]=f2b(v); }
  __syncthreads();

  // stage1: T(32x64) = Dh32(32x64) x X(64x64); 8 waves = 2x4 tiles, K=64
  {
    int mi=wave>>2, ni=wave&3;
    v8f acc={};
    #pragma unroll
    for(int k0=0;k0<64;k0+=32){
      v16bf a,b;
      const bf16* ar=dhb+(size_t)(mi*16+l15)*64+k0;
      bf16x8 a0=*(const bf16x8*)(ar+kh*16), a1=*(const bf16x8*)(ar+kh*16+8);
      #pragma unroll
      for(int i=0;i<8;i++){ a[i]=a0[i]; a[i+8]=a1[i]; }
      const bf16* br=Xt+(size_t)(ni*16+l15)*64+k0+kh*16;
      #pragma unroll
      for(int i=0;i<16;i++) b[i]=br[i];
      acc=__builtin_amdgcn_wmma_f32_16x16x32_bf16(false,a,false,b,(short)0,acc,false,false);
    }
    #pragma unroll
    for(int v=0;v<8;v++) T[(mi*16+v+kh*8)*64 + ni*16+l15]=f2b(acc[v]);
  }
  __syncthreads();

  // stage2: F(32x32) = T(32x64) x Dh32^T; waves 0..3 = 2x2 tiles; store transposed Ft[l][k]
  if(wave<4){
    int mi=wave>>1, ni=wave&1;
    v8f acc={};
    #pragma unroll
    for(int k0=0;k0<64;k0+=32){
      v16bf a,b;
      const bf16* ar=T+(size_t)(mi*16+l15)*64+k0;
      #pragma unroll
      for(int i=0;i<8;i++){ a[i]=ar[kh*16+i]; a[i+8]=ar[kh*16+8+i]; }
      const bf16* br=dhb+(size_t)(ni*16+l15)*64+k0+kh*16;  // B[w][l]=dh[l][w]: row l of dhb
      #pragma unroll
      for(int i=0;i<16;i++) b[i]=br[i];
      acc=__builtin_amdgcn_wmma_f32_16x16x32_bf16(false,a,false,b,(short)0,acc,false,false);
    }
    #pragma unroll
    for(int v=0;v<8;v++) Ft[(ni*16+l15)*32 + mi*16+v+kh*8]=f2b(acc[v]);
  }
  __syncthreads();

  // stage3: U(64x32) = Dh32^T(64x32) x F(32x32); 8 waves = 4x2 tiles, K=32
  {
    int mi=wave>>1, ni=wave&1;
    v8f acc={};
    v16bf a,b;
    const bf16* ar=dhtb+(size_t)(mi*16+l15)*32;
    bf16x8 a0=*(const bf16x8*)(ar+kh*16), a1=*(const bf16x8*)(ar+kh*16+8);
    #pragma unroll
    for(int i=0;i<8;i++){ a[i]=a0[i]; a[i+8]=a1[i]; }
    const bf16* br=Ft+(size_t)(ni*16+l15)*32+kh*16;
    #pragma unroll
    for(int i=0;i<16;i++) b[i]=br[i];
    acc=__builtin_amdgcn_wmma_f32_16x16x32_bf16(false,a,false,b,(short)0,acc,false,false);
    #pragma unroll
    for(int v=0;v<8;v++) U[(mi*16+v+kh*8)*32 + ni*16+l15]=f2b(acc[v]);
  }
  __syncthreads();

  // stage4: low(64x64) = U(64x32) x Dh32(32x64); 16 tiles, 2 per wave; high = X - low
  #pragma unroll
  for(int t2=0;t2<2;t2++){
    int id=wave*2+t2, mi=id>>2, ni=id&3;
    v8f acc={};
    v16bf a,b;
    const bf16* ar=U+(size_t)(mi*16+l15)*32;
    #pragma unroll
    for(int i=0;i<8;i++){ a[i]=ar[kh*16+i]; a[i+8]=ar[kh*16+8+i]; }
    const bf16* br=dhtb+(size_t)(ni*16+l15)*32+kh*16;     // B[l][w]=dh[l][w]: dhtb[w][l]
    #pragma unroll
    for(int i=0;i<16;i++) b[i]=br[i];
    acc=__builtin_amdgcn_wmma_f32_16x16x32_bf16(false,a,false,b,(short)0,acc,false,false);
    int wcol=ni*16+l15;
    #pragma unroll
    for(int v=0;v<8;v++){
      int hrow=mi*16+v+kh*8;
      float hv=Xf[hrow*64+wcol]-acc[v];
      high_in[(((size_t)bb_<<12)+(hrow*64+wcol))*256 + cc]=f2b(hv);
    }
  }
  __syncthreads();

  // stage5: lowp = Ph F Ph^T (tiny, VALU)
  for(int i=tid;i<8*32;i+=256){
    int p=i>>5,l=i&31; float a=0.f;
    for(int k=0;k<32;k++) a+=ph[p*32+k]*b2f(Ft[l*32+k]);
    V[i]=a;
  }
  __syncthreads();
  for(int i=tid;i<64;i+=256){
    int p=i>>3,q=i&7; float a=0.f;
    for(int l=0;l<32;l++) a+=V[p*32+l]*ph[q*32+l];
    lowp[(((size_t)bb_*64)+i)*256 + cc]=f2b(a);
  }
}

// ---------------------------------------------------------------- implicit-GEMM conv via WMMA (NHWC, LDS-free)
// X NHWC [B][H*W][Cin] bf16; Wt [Cout][Ktot] (3x3: k = tap*Cin+ci)
__global__ void __launch_bounds__(256) k_gemm(
    const bf16* __restrict__ X, const bf16* __restrict__ Wt, const float* __restrict__ bias,
    float* __restrict__ outF, int outF_nchw,
    bf16* __restrict__ outH, int outHC, int choff,
    const float* __restrict__ res /*NCHW*/,
    int Cin,int Cout,int H,int W,int khw)
{
  const int HW=H*W, Ktot=Cin*khw;
  const int b=blockIdx.z, co0=blockIdx.y*32, n0=blockIdx.x*64;
  const int tid=threadIdx.x, lane=tid&31, wave=tid>>5;
  const int mi=wave>>2, ni=wave&3, l15=lane&15, kh=lane>>4;

  const int n=n0+ni*16+l15;                 // this lane's B/D column (pixel)
  const int h=n/W, w=n-(n/W)*W;
  const int arow=co0+mi*16+l15;             // this lane's A row (out channel)
  const bf16* Xb=X+(size_t)b*HW*Cin;
  const bf16* Ar=Wt+(size_t)arow*Ktot;

  v8f acc={};
  for(int k0=0;k0<Ktot;k0+=32){
    if(k0+32<Ktot) __builtin_prefetch(Ar+k0+32,0,1);
    v16bf a,bb;
    bf16x8 a0=*(const bf16x8*)(Ar+k0+kh*16), a1=*(const bf16x8*)(Ar+k0+kh*16+8);
    #pragma unroll
    for(int i=0;i<8;i++){ a[i]=a0[i]; a[i+8]=a1[i]; }
    int ci0=k0+kh*16, hh=h, ww=w;
    if(khw==9){ int tap=ci0/Cin; ci0-=tap*Cin; int ty=tap/3; hh=h+ty-1; ww=w+(tap-ty*3)-1; }
    #pragma unroll
    for(int i=0;i<16;i++) bb[i]=(bf16)0.0f;
    if(hh>=0&&hh<H&&ww>=0&&ww<W) bb=*(const v16bf*)(Xb+(size_t)(hh*W+ww)*Cin+ci0);
    acc=__builtin_amdgcn_wmma_f32_16x16x32_bf16(false,a,false,bb,(short)0,acc,false,false);
  }

  #pragma unroll
  for(int v=0;v<8;v++){
    int co=co0+mi*16+v+kh*8;
    float val=acc[v]+bias[co];
    if(outF_nchw){
      size_t oi=((size_t)b*Cout+co)*HW+n;
      if(res) val+=res[oi];
      outF[oi]=val;
    }else if(outF){
      outF[((size_t)b*HW+n)*Cout+co]=val;
    }
    if(outH) outH[((size_t)b*HW+n)*outHC+choff+co]=f2b(val);
  }
}

// ---------------------------------------------------------------- GroupNorm(8) stats, NHWC f32 input
__global__ void __launch_bounds__(256) k_gn_stats(const float* __restrict__ x, float* __restrict__ stats,
                                                  int C,int HW){
  int bg=blockIdx.x, b=bg>>3, g=bg&7;
  const float* base=x+(size_t)b*HW*C+g*32;
  long cnt=(long)HW*32;
  float s=0.f,s2=0.f;
  for(long i=threadIdx.x;i<cnt;i+=256){
    int j=(int)(i&31); long hw=i>>5;
    float v=base[hw*C+j]; s+=v; s2+=v*v;
  }
  __shared__ float rs[256],rq[256];
  rs[threadIdx.x]=s; rq[threadIdx.x]=s2; __syncthreads();
  for(int o=128;o>0;o>>=1){
    if(threadIdx.x<o){ rs[threadIdx.x]+=rs[threadIdx.x+o]; rq[threadIdx.x]+=rq[threadIdx.x+o]; }
    __syncthreads();
  }
  if(threadIdx.x==0){
    float mean=rs[0]/(float)cnt, var=rq[0]/(float)cnt-mean*mean;
    stats[bg*2]=mean; stats[bg*2+1]=rsqrtf(var+1e-5f);
  }
}

// ---------------------------------------------------------------- GN affine + exact GELU, NHWC f32 -> NHWC bf16 (+channel offset)
__global__ void __launch_bounds__(256) k_gn_apply(const float* __restrict__ x, const float* __restrict__ stats,
                                                  const float* __restrict__ gamma, const float* __restrict__ beta,
                                                  bf16* __restrict__ out, int hwsh, int outC, int choff, long n){
  long i=(long)blockIdx.x*256+threadIdx.x;
  if(i>=n) return;
  int c=(int)(i&255); long t=i>>8;
  long hw=t&((1L<<hwsh)-1); long b=t>>hwsh;
  int g=c>>5;
  float mean=stats[(b*8+g)*2], inv=stats[(b*8+g)*2+1];
  float v=(x[i]-mean)*inv*gamma[c]+beta[c];
  out[(((size_t)b<<hwsh)+hw)*outC+choff+c]=f2b(gelu_exact(v));
}

// ---------------------------------------------------------------- depthwise 3x3 pad1, NHWC bf16 (64x64, C=256)
__global__ void __launch_bounds__(256) k_dw3x3(const bf16* __restrict__ x, const float* __restrict__ wgt,
                                               const float* __restrict__ bias, bf16* __restrict__ out, long n){
  long i=(long)blockIdx.x*256+threadIdx.x;
  if(i>=n) return;
  int c=(int)(i&255); long pix=(i>>8)&4095; long b=i>>20;
  int h=(int)(pix>>6), w=(int)(pix&63);
  const bf16* xb=x+((size_t)b<<12)*256;
  float a=bias[c];
  #pragma unroll
  for(int dy=-1;dy<=1;dy++)
    #pragma unroll
    for(int dx=-1;dx<=1;dx++){
      int hh=h+dy, wc=w+dx;
      if(hh>=0&&hh<64&&wc>=0&&wc<64)
        a+=wgt[c*9+(dy+1)*3+(dx+1)]*b2f(xb[(size_t)(hh*64+wc)*256+c]);
    }
  out[i]=f2b(a);
}

// ---------------------------------------------------------------- bilinear 8->64 (align_corners) -> cat channels 512..767 (NHWC)
__global__ void __launch_bounds__(256) k_upsample(const float* __restrict__ low8 /*NHWC [b][64][256]*/,
                                                  bf16* __restrict__ cat, long n){
  long i=(long)blockIdx.x*256+threadIdx.x;
  if(i>=n) return;
  int c=(int)(i&255); long pix=(i>>8)&4095; long b=i>>20;
  int ho=(int)(pix>>6), wo=(int)(pix&63);
  float fy=ho*(7.0f/63.0f); int y0=(int)floorf(fy); int y1=(y0+1<7)?y0+1:7; float ty=fy-(float)y0;
  float fx=wo*(7.0f/63.0f); int x0=(int)floorf(fx); int x1=(x0+1<7)?x0+1:7; float tx=fx-(float)x0;
  const float* p=low8+(size_t)b*64*256;
  float v=(1.f-ty)*((1.f-tx)*p[(y0*8+x0)*256+c]+tx*p[(y0*8+x1)*256+c])
        +      ty *((1.f-tx)*p[(y1*8+x0)*256+c]+tx*p[(y1*8+x1)*256+c]);
  cat[(((size_t)b<<12)+pix)*768+512+c]=f2b(v);
}

// ================================================================ host
extern "C" void kernel_launch(void* const* d_in, const int* in_sizes, int n_in,
                              void* d_out, int out_size, void* d_ws, size_t ws_size,
                              hipStream_t stream){
  const float* x1      =(const float*)d_in[0];
  const float* x2      =(const float*)d_in[1];
  const float* s2f_w   =(const float*)d_in[2];
  const float* s2f_b   =(const float*)d_in[3];
  const float* s2f_g   =(const float*)d_in[4];
  const float* s2f_bt  =(const float*)d_in[5];
  // d_in[6..9]: attention params -> dead code in reference, skipped
  const float* hfe_dw1  =(const float*)d_in[10];
  const float* hfe_dw1b =(const float*)d_in[11];
  const float* hfe_pw1  =(const float*)d_in[12];
  const float* hfe_pw1b =(const float*)d_in[13];
  const float* hfe_g    =(const float*)d_in[14];
  const float* hfe_bt   =(const float*)d_in[15];
  const float* hfe_dw2  =(const float*)d_in[16];
  const float* hfe_dw2b =(const float*)d_in[17];
  const float* hfe_pw2  =(const float*)d_in[18];
  const float* hfe_pw2b =(const float*)d_in[19];
  const float* lfs_w1   =(const float*)d_in[20];
  const float* lfs_b1   =(const float*)d_in[21];
  const float* lfs_g    =(const float*)d_in[22];
  const float* lfs_bt   =(const float*)d_in[23];
  const float* lfs_w2   =(const float*)d_in[24];
  const float* lfs_b2   =(const float*)d_in[25];
  const float* fus_w1   =(const float*)d_in[26];
  const float* fus_b1   =(const float*)d_in[27];
  const float* fus_g    =(const float*)d_in[28];
  const float* fus_bt   =(const float*)d_in[29];
  const float* fus_w2   =(const float*)d_in[30];
  const float* fus_b2   =(const float*)d_in[31];
  float* out=(float*)d_out;

  const long N = 8L*256*64*64;           // 8,388,608
  const long NL= 8L*256*64;

  char* ws=(char*)d_ws; size_t off=0;
  auto alloc=[&](size_t bytes)->char*{ char* p=ws+off; off+=(bytes+255)&~(size_t)255; return p; };

  bf16*  dhb      =(bf16*)alloc(32*64*2);
  bf16*  dhtb     =(bf16*)alloc(64*32*2);
  float* ph       =(float*)alloc(8*32*4);
  float* diff_f   =(float*)alloc(N*4);        // NCHW (residual)
  bf16*  diff_b   =(bf16*)alloc(N*2);         // NHWC
  bf16*  high_b   =(bf16*)alloc(N*2);         // NHWC: high_in, later dw2 output
  bf16*  buf_b    =(bf16*)alloc(N*2);         // NHWC: dw1 out / GN outputs
  bf16*  lowp_b   =(bf16*)alloc(NL*2);        // NHWC [b][64][256]
  float* tmpA     =(float*)alloc(N*4);        // NHWC f32 pre-GN
  bf16*  cat_b    =(bf16*)alloc(8L*4096*768*2); // NHWC [b][4096][768]
  float* lc1      =(float*)alloc(NL*4);
  bf16*  lcmid    =(bf16*)alloc(NL*2);
  float* lc2      =(float*)alloc(NL*4);
  float* stats    =(float*)alloc(64*2*4);
  bf16*  w_s2f    =(bf16*)alloc(256*256*2);
  bf16*  w_pw1    =(bf16*)alloc(256*256*2);
  bf16*  w_pw2    =(bf16*)alloc(256*256*2);
  bf16*  w_lfs1   =(bf16*)alloc(256*2304*2);
  bf16*  w_lfs2   =(bf16*)alloc(256*2304*2);
  bf16*  w_fus1   =(bf16*)alloc(256*768*2);
  bf16*  w_fus2   =(bf16*)alloc(256*2304*2);

  dim3 blk(256);
  auto gfor=[&](long n){ return dim3((unsigned)((n+255)/256)); };

  // 0) basis + weight conversion + diff
  k_basis<<<1,blk,0,stream>>>(dhb,dhtb,ph);
  k_f2b  <<<gfor(65536),blk,0,stream>>>(s2f_w,  w_s2f, 65536);
  k_f2b  <<<gfor(65536),blk,0,stream>>>(hfe_pw1,w_pw1, 65536);
  k_f2b  <<<gfor(65536),blk,0,stream>>>(hfe_pw2,w_pw2, 65536);
  k_f2b  <<<gfor(196608),blk,0,stream>>>(fus_w1,w_fus1,196608);
  k_wperm<<<gfor(589824),blk,0,stream>>>(lfs_w1,w_lfs1,589824);
  k_wperm<<<gfor(589824),blk,0,stream>>>(lfs_w2,w_lfs2,589824);
  k_wperm<<<gfor(589824),blk,0,stream>>>(fus_w2,w_fus2,589824);
  k_diff <<<gfor(N),blk,0,stream>>>(x1,x2,diff_f,diff_b,N);

  // 1) DCT low-block (WMMA): high_in + lowp
  k_dct<<<2048,blk,0,stream>>>(diff_f,dhb,dhtb,ph,high_b,lowp_b);

  // 2) spatial branch: 1x1 conv -> GN -> GELU -> cat[:, :,0:256]
  k_gemm<<<dim3(64,8,8),blk,0,stream>>>(diff_b,w_s2f,s2f_b,tmpA,0,(bf16*)nullptr,0,0,(const float*)nullptr,
                                        256,256,64,64,1);
  k_gn_stats<<<64,blk,0,stream>>>(tmpA,stats,256,4096);
  k_gn_apply<<<gfor(N),blk,0,stream>>>(tmpA,stats,s2f_g,s2f_bt,cat_b,12,768,0,N);

  // 3) high branch: dw1 -> pw1 -> GN+GELU -> dw2 -> pw2 -> cat[:, :,256:512]
  k_dw3x3<<<gfor(N),blk,0,stream>>>(high_b,hfe_dw1,hfe_dw1b,buf_b,N);
  k_gemm<<<dim3(64,8,8),blk,0,stream>>>(buf_b,w_pw1,hfe_pw1b,tmpA,0,(bf16*)nullptr,0,0,(const float*)nullptr,
                                        256,256,64,64,1);
  k_gn_stats<<<64,blk,0,stream>>>(tmpA,stats,256,4096);
  k_gn_apply<<<gfor(N),blk,0,stream>>>(tmpA,stats,hfe_g,hfe_bt,buf_b,12,256,0,N);
  k_dw3x3<<<gfor(N),blk,0,stream>>>(buf_b,hfe_dw2,hfe_dw2b,high_b,N);
  k_gemm<<<dim3(64,8,8),blk,0,stream>>>(high_b,w_pw2,hfe_pw2b,(float*)nullptr,0,cat_b,768,256,(const float*)nullptr,
                                        256,256,64,64,1);

  // 4) low branch on 8x8: conv3x3 -> GN+GELU -> conv3x3 -> bilinear up -> cat[:, :,512:768]
  k_gemm<<<dim3(1,8,8),blk,0,stream>>>(lowp_b,w_lfs1,lfs_b1,lc1,0,(bf16*)nullptr,0,0,(const float*)nullptr,
                                       256,256,8,8,9);
  k_gn_stats<<<64,blk,0,stream>>>(lc1,stats,256,64);
  k_gn_apply<<<gfor(NL),blk,0,stream>>>(lc1,stats,lfs_g,lfs_bt,lcmid,6,256,0,NL);
  k_gemm<<<dim3(1,8,8),blk,0,stream>>>(lcmid,w_lfs2,lfs_b2,lc2,0,(bf16*)nullptr,0,0,(const float*)nullptr,
                                       256,256,8,8,9);
  k_upsample<<<gfor(N),blk,0,stream>>>(lc2,cat_b,N);

  // 5) fusion: 1x1 (768->256) -> GN+GELU -> 3x3 + bias + residual(diff) -> d_out (NCHW)
  k_gemm<<<dim3(64,8,8),blk,0,stream>>>(cat_b,w_fus1,fus_b1,tmpA,0,(bf16*)nullptr,0,0,(const float*)nullptr,
                                        768,256,64,64,1);
  k_gn_stats<<<64,blk,0,stream>>>(tmpA,stats,256,4096);
  k_gn_apply<<<gfor(N),blk,0,stream>>>(tmpA,stats,fus_g,fus_bt,buf_b,12,256,0,N);
  k_gemm<<<dim3(64,8,8),blk,0,stream>>>(buf_b,w_fus2,fus_b2,out,1,(bf16*)nullptr,0,0,diff_f,
                                        256,256,64,64,9);

  (void)in_sizes;(void)n_in;(void)out_size;(void)ws_size;
}
